// ConditionalSetEncoder_55327768707115
// MI455X (gfx1250) — matrix operand order repaired and checked by reading, not compiled, not run on hardware
//
#include <hip/hip_runtime.h>

// ============================================================================
// ConditionalSetEncoder for MI455X (gfx1250, wave32, WMMA bf16).
//
// GEMMs: fused-epilogue bf16 WMMA, f32 accumulation. The 64xK weight tile is
// staged into LDS once per block with GLOBAL_LOAD_ASYNC_TO_LDS_B128 (ASYNCcnt
// path) and shared by all 8 waves; A rows stream from global/L2.
// Attention: per-(batch,head,32-q-rows) block, WMMA logits -> LDS softmax ->
// WMMA P@V^T with fused q-residual.
//
// Scratch budget (~549 MiB) assumed available in d_ws.
// ============================================================================

#define BATCH 64
#define NC    1024
#define ND    1024
#define DXCH  64      // DX
#define DCCH  128     // DX+DY
#define DH    512
#define NHEAD 8
#define HD    64
#define NROW  (BATCH*NC)   // 65536 rows (== BATCH*ND)

typedef __attribute__((ext_vector_type(16))) __bf16 v16bf;
typedef __attribute__((ext_vector_type(8)))  __bf16 v8bf;
typedef __attribute__((ext_vector_type(8)))  float  v8f;

static __device__ __forceinline__ v8bf  ld8 (const __bf16* p){ return *(const v8bf *)p; }
static __device__ __forceinline__ v16bf ld16(const __bf16* p){ return *(const v16bf*)p; }
static __device__ __forceinline__ v16bf comb(v8bf lo, v8bf hi){
  return __builtin_shufflevector(lo, hi, 0,1,2,3,4,5,6,7,8,9,10,11,12,13,14,15);
}
static __device__ __forceinline__ v8f wmma32(v16bf a, v16bf b, v8f c){
  // v_wmma_f32_16x16x32_bf16: D = A(16x32) * B(32x16) + C
  return __builtin_amdgcn_wmma_f32_16x16x32_bf16(false, a, false, b, (short)0, c, false, false);
}
// A operand: lane&15 = M row; half = lane>>4 selects contiguous K block h*16
static __device__ __forceinline__ v16bf loadA(const __bf16* row, int k0, int half){
  return comb(ld8(row + k0 + half*16), ld8(row + k0 + half*16 + 8));
}
// B operand: lane&15 = N row of Wt[N][K]; half selects contiguous K block h*16
static __device__ __forceinline__ v16bf loadB(const __bf16* row, int k0, int half){
  return comb(ld8(row + k0 + half*16), ld8(row + k0 + half*16 + 8));
}

// ---- CDNA5 async copy: global -> LDS, 16B per lane-op, tracked by ASYNCcnt --
static __device__ __forceinline__ void async_g2l_b128(unsigned lds_byte_addr,
                                                      const void* gptr){
  asm volatile("global_load_async_to_lds_b128 %0, %1, off"
               :: "v"(lds_byte_addr), "v"(gptr) : "memory");
}
static __device__ __forceinline__ void wait_asynccnt0(){
  asm volatile("s_wait_asynccnt 0x0" ::: "memory");
}

// ---------------------------------------------------------------------------
// prep kernels
// ---------------------------------------------------------------------------
__global__ void k_mask(const float* __restrict__ C, float* __restrict__ maskF){
  int i = blockIdx.x*256 + threadIdx.x;
  if (i < NROW){ float x = C[(long)i*DCCH + (DCCH-1)]; maskF[i] = (x != x) ? 1.f : 0.f; }
}
__global__ void k_cvt_nz(const float* __restrict__ s, __bf16* __restrict__ d, long n){
  long i = (long)blockIdx.x*256 + threadIdx.x;
  if (i < n){ float x = s[i]; if (x != x) x = 0.f; d[i] = (__bf16)x; }
}
__global__ void k_cvt(const float* __restrict__ s, __bf16* __restrict__ d, long n){
  long i = (long)blockIdx.x*256 + threadIdx.x;
  if (i < n) d[i] = (__bf16)s[i];
}
// transpose-convert weight: src f32 [K,N] row-major -> dst bf16 [N,K]
__global__ void k_wt(const float* __restrict__ src, __bf16* __restrict__ dst, int Kd, int Nd){
  int i = blockIdx.x*256 + threadIdx.x;
  if (i < Kd*Nd){ int k = i / Nd, n = i % Nd; dst[(long)n*Kd + k] = (__bf16)src[i]; }
}

// ---------------------------------------------------------------------------
// fused WMMA GEMM:  out[M,512] = epilogue( A[M,K](bf16) @ Wt[512,K]^T )
// wave tile 16(M) x 64(N); block = 8 waves -> 128(M) x 64(N).
// The block's 64xK weight tile is async-DMA'd into LDS once and reused by all
// 8 waves (ds_load path); A rows stream from global (L2-resident activations).
// Dynamic LDS = 64*K*2 bytes (8..64 KB).
// ---------------------------------------------------------------------------
template<int K, bool BIAS, bool RELU, bool MASKTASK, bool RESID,
         bool STF, bool STB, bool STT>
__global__ __launch_bounds__(256) void gemm_k(
    const __bf16* __restrict__ A, const __bf16* __restrict__ Wt,
    const float* __restrict__ bias, const float* __restrict__ resid,
    const float* __restrict__ maskF, const float* __restrict__ taskEmb,
    float* __restrict__ outF, __bf16* __restrict__ outB, __bf16* __restrict__ outT)
{
  extern __shared__ char gsm[];
  __bf16* sW = (__bf16*)gsm;                     // [64][K]

  const int tid  = threadIdx.x;
  const int wave = tid >> 5;
  const int lane = tid & 31;
  const int r15 = lane & 15, half = lane >> 4;
  const long m0 = (long)blockIdx.y*128 + wave*16;
  const int  n0 = blockIdx.x*64;

  // ---- async stage: Wt rows [n0, n0+64) x K  ->  LDS (contiguous stream) ----
  {
    const char* src = (const char*)(Wt + (long)n0*K);
    const unsigned sBase = (unsigned)(unsigned long long)(const void*)sW;
    const int chunks = (64*K*2)/16;              // 16-byte chunks
    for (int i = tid; i < chunks; i += 256)
      async_g2l_b128(sBase + i*16, src + i*16);
    wait_asynccnt0();
    __syncthreads();
  }

  const __bf16* arow = A + (m0 + r15)*K;

  v8f acc[4] = {};
  #pragma unroll 4
  for (int k0 = 0; k0 < K; k0 += 32){
    v16bf a = loadA(arow, k0, half);
    #pragma unroll
    for (int j = 0; j < 4; j++)
      acc[j] = wmma32(a, loadB(sW + (j*16 + r15)*K, k0, half), acc[j]);
  }

  // C layout: VGPR r holds rows r (lanes 0-15) and r+8 (lanes 16-31), col = lane&15
  #pragma unroll
  for (int r = 0; r < 8; r++){
    const long m = m0 + half*8 + r;
    float mflag = 0.f;
    if (MASKTASK) mflag = maskF[m];
    #pragma unroll
    for (int j = 0; j < 4; j++){
      const int n = n0 + j*16 + r15;
      float v = acc[j][r];
      if (BIAS) v += bias[n];
      if (RELU) v = fmaxf(v, 0.f);
      if (MASKTASK){ if (mflag != 0.f) v = 0.f; v += taskEmb[n]; }
      if (RESID) v += resid[m*DH + n];
      if (STF) outF[m*DH + n] = v;
      if (STB) outB[m*DH + n] = (__bf16)v;
      if (STT){ // transposed store: vT[b][n][key], key = m % NC
        const long bb = m >> 10; const int key = (int)(m & (NC-1));
        outT[(bb*DH + n)*NC + key] = (__bf16)v;
      }
    }
  }
}

// ---------------------------------------------------------------------------
// cross-attention: one block = (batch b, head h, 32 query rows)
//   phase1: logits = (q @ k^T)/sqrt(DH), mask -> LDS (32x1024 f32)
//   phase2: row softmax -> probs bf16 (32x1024)
//   phase3: o = P @ V^T(vT);  qF += o (residual, in place);  outB = bf16(qF)
// Each K/V row is read by only 2 waves of a block (one per q tile), so they stream directly.
// ---------------------------------------------------------------------------
#define ATTN_SMEM (32*1024*4 + 32*1024*2 + 32*8*4)   // 197632 B < 320KB/WGP

__global__ __launch_bounds__(256) void attn_k(
    const __bf16* __restrict__ qB, const __bf16* __restrict__ kB,
    const __bf16* __restrict__ vT, const float* __restrict__ maskF,
    float* __restrict__ qF, __bf16* __restrict__ outB)
{
  extern __shared__ char smem[];
  float*  sLog  = (float*)smem;                              // [32][1024]
  __bf16* sProb = (__bf16*)(smem + 32*1024*4);               // [32][1024]
  float*  sRed  = (float*)(smem + 32*1024*4 + 32*1024*2);    // [32][8]

  const int b = blockIdx.z, h = blockIdx.y, q0 = blockIdx.x*32;
  const int tid = threadIdx.x, wave = tid >> 5, lane = tid & 31;
  const int r15 = lane & 15, half = lane >> 4;
  const float scale = 0.04419417382415922f;  // 1/sqrt(512)

  { // ---- phase 1: logits ----
    const int qt = wave & 1;                       // which 16-row q tile
    const long qrow = (long)b*ND + q0 + qt*16 + r15;
    const __bf16* qp = qB + qrow*DH + h*HD;
    v16bf a0 = comb(ld8(qp +      half*16), ld8(qp +  8 + half*16));
    v16bf a1 = comb(ld8(qp + 32 + half*16), ld8(qp + 40 + half*16));
    const int kb = (wave >> 1)*256;                // 256 keys per wave
    for (int t = 0; t < 16; t++){
      const int k0 = kb + t*16;
      const __bf16* kp = kB + ((long)b*NC + k0 + r15)*DH + h*HD;
      v8f c = {};
      c = wmma32(a0, ld16(kp +      half*16), c);
      c = wmma32(a1, ld16(kp + 32 + half*16), c);
      const float mflag = maskF[b*NC + k0 + r15];
      #pragma unroll
      for (int r = 0; r < 8; r++){
        float v = c[r]*scale;
        if (mflag != 0.f) v = -__builtin_inff();
        sLog[(qt*16 + half*8 + r)*1024 + (k0 + r15)] = v;
      }
    }
  }
  __syncthreads();

  { // ---- phase 2: softmax over keys, 8 threads per row ----
    const int row = tid >> 3, sub = tid & 7;
    float* lrow = sLog + row*1024;
    float mx = -__builtin_inff();
    for (int k = sub; k < 1024; k += 8) mx = fmaxf(mx, lrow[k]);
    sRed[row*8 + sub] = mx;
    __syncthreads();
    float rmax = sRed[row*8];
    #pragma unroll
    for (int i = 1; i < 8; i++) rmax = fmaxf(rmax, sRed[row*8 + i]);
    __syncthreads();
    float s = 0.f;
    for (int k = sub; k < 1024; k += 8){
      float e = __expf(lrow[k] - rmax); lrow[k] = e; s += e;
    }
    sRed[row*8 + sub] = s;
    __syncthreads();
    float rs = 0.f;
    #pragma unroll
    for (int i = 0; i < 8; i++) rs += sRed[row*8 + i];
    const float inv = 1.f / rs;
    for (int k = sub; k < 1024; k += 8)
      sProb[row*1024 + k] = (__bf16)(lrow[k]*inv);
  }
  __syncthreads();

  { // ---- phase 3: O = P @ V^T, fused residual ----
    const int qt = wave >> 2, dt = wave & 3;       // 2 q tiles x 4 dim tiles
    const __bf16* pr = sProb + (qt*16 + r15)*1024;
    const __bf16* vr = vT + ((long)b*DH + h*HD + dt*16 + r15)*NC;
    v8f c = {};
    for (int kc = 0; kc < 1024; kc += 32){
      v16bf a = comb(ld8(pr + kc + half*16), ld8(pr + kc + 8 + half*16));
      c = wmma32(a, ld16(vr + kc + half*16), c);
    }
    #pragma unroll
    for (int r = 0; r < 8; r++){
      const long m = (long)b*ND + q0 + qt*16 + half*8 + r;
      const int  n = h*HD + dt*16 + r15;
      float v = qF[m*DH + n] + c[r];
      qF[m*DH + n] = v;                 // out = q + o   (f32, FFN residual)
      outB[m*DH + n] = (__bf16)v;       // bf16 copy for Wo GEMM
    }
  }
}

// ---------------------------------------------------------------------------
// host orchestration
// ---------------------------------------------------------------------------
extern "C" void kernel_launch(void* const* d_in, const int* in_sizes, int n_in,
                              void* d_out, int out_size, void* d_ws, size_t ws_size,
                              hipStream_t stream)
{
  (void)in_sizes; (void)n_in; (void)out_size; (void)ws_size;
  const float* C   = (const float*)d_in[0];
  const float* XC  = (const float*)d_in[1];
  const float* XD  = (const float*)d_in[2];
  const float* W0  = (const float*)d_in[3];
  const float* b0  = (const float*)d_in[4];
  const float* Wm  = (const float*)d_in[5];   // [2,512,512]
  const float* bm  = (const float*)d_in[6];   // [2,512]
  const float* Wl  = (const float*)d_in[7];
  const float* bl  = (const float*)d_in[8];
  const float* te  = (const float*)d_in[9];
  const float* Wq0 = (const float*)d_in[10];
  const float* Wk0 = (const float*)d_in[11];
  const float* Wv0 = (const float*)d_in[12];
  const float* Wo0 = (const float*)d_in[13];
  const float* bo0 = (const float*)d_in[14];
  const float* Wq1 = (const float*)d_in[15];
  const float* Wk1 = (const float*)d_in[16];
  const float* Wv1 = (const float*)d_in[17];
  const float* Wo1 = (const float*)d_in[18];
  const float* bo1 = (const float*)d_in[19];

  char* ws = (char*)d_ws;
  size_t off = 0;
  auto take = [&](size_t bytes)->char*{
    char* p = ws + off; off += (bytes + 255) & ~(size_t)255; return p;
  };
  float*  maskF = (float*) take((size_t)NROW*4);
  __bf16* W0t   = (__bf16*)take((size_t)DH*DCCH*2);
  __bf16* Wm0t  = (__bf16*)take((size_t)DH*DH*2);
  __bf16* Wm1t  = (__bf16*)take((size_t)DH*DH*2);
  __bf16* Wlt   = (__bf16*)take((size_t)DH*DH*2);
  __bf16* Wq0t  = (__bf16*)take((size_t)DH*DXCH*2);
  __bf16* Wk0t  = (__bf16*)take((size_t)DH*DXCH*2);
  __bf16* Wv0t  = (__bf16*)take((size_t)DH*DH*2);
  __bf16* Wo0t  = (__bf16*)take((size_t)DH*DH*2);
  __bf16* Wq1t  = (__bf16*)take((size_t)DH*DH*2);
  __bf16* Wk1t  = (__bf16*)take((size_t)DH*DXCH*2);
  __bf16* Wv1t  = (__bf16*)take((size_t)DH*DH*2);
  __bf16* Wo1t  = (__bf16*)take((size_t)DH*DH*2);
  __bf16* Czb   = (__bf16*)take((size_t)NROW*DCCH*2);
  __bf16* XCb   = (__bf16*)take((size_t)NROW*DXCH*2);
  __bf16* XDb   = (__bf16*)take((size_t)NROW*DXCH*2);
  __bf16* hA    = (__bf16*)take((size_t)NROW*DH*2);
  __bf16* hB    = (__bf16*)take((size_t)NROW*DH*2);
  __bf16* dbf   = (__bf16*)take((size_t)NROW*DH*2);
  __bf16* qbf   = (__bf16*)take((size_t)NROW*DH*2);
  __bf16* kbf   = (__bf16*)take((size_t)NROW*DH*2);
  __bf16* vT    = (__bf16*)take((size_t)BATCH*DH*NC*2);
  float*  qF    = (float*) take((size_t)NROW*DH*4);

  // ---- conversions ----
  k_mask<<<(NROW+255)/256, 256, 0, stream>>>(C, maskF);
  k_cvt_nz<<<(unsigned)(((long)NROW*DCCH+255)/256), 256, 0, stream>>>(C, Czb, (long)NROW*DCCH);
  k_cvt  <<<(unsigned)(((long)NROW*DXCH+255)/256), 256, 0, stream>>>(XC, XCb, (long)NROW*DXCH);
  k_cvt  <<<(unsigned)(((long)NROW*DXCH+255)/256), 256, 0, stream>>>(XD, XDb, (long)NROW*DXCH);
  const int gb5 = (DH*DH+255)/256, gb1 = (DCCH*DH+255)/256, gb0 = (DXCH*DH+255)/256;
  k_wt<<<gb1,256,0,stream>>>(W0,        W0t,  DCCH, DH);
  k_wt<<<gb5,256,0,stream>>>(Wm,        Wm0t, DH,   DH);
  k_wt<<<gb5,256,0,stream>>>(Wm+DH*DH,  Wm1t, DH,   DH);
  k_wt<<<gb5,256,0,stream>>>(Wl,        Wlt,  DH,   DH);
  k_wt<<<gb0,256,0,stream>>>(Wq0,       Wq0t, DXCH, DH);
  k_wt<<<gb0,256,0,stream>>>(Wk0,       Wk0t, DXCH, DH);
  k_wt<<<gb5,256,0,stream>>>(Wv0,       Wv0t, DH,   DH);
  k_wt<<<gb5,256,0,stream>>>(Wo0,       Wo0t, DH,   DH);
  k_wt<<<gb5,256,0,stream>>>(Wq1,       Wq1t, DH,   DH);
  k_wt<<<gb0,256,0,stream>>>(Wk1,       Wk1t, DXCH, DH);
  k_wt<<<gb5,256,0,stream>>>(Wv1,       Wv1t, DH,   DH);
  k_wt<<<gb5,256,0,stream>>>(Wo1,       Wo1t, DH,   DH);

  const dim3 gg(DH/64, NROW/128), bb(256);
  const dim3 ga(ND/32, NHEAD, BATCH);
  const size_t smK512 = (size_t)64*DH*2;    // 64 KB weight tile
  const size_t smK128 = (size_t)64*DCCH*2;  // 16 KB
  const size_t smK64  = (size_t)64*DXCH*2;  //  8 KB

  // ---- MLP: h = relu(Cz@W0+b0) -> relu(@Wm0+bm0) -> relu(@Wm1+bm1) -> d ----
  gemm_k<DCCH, true,true,false,false, false,true,false><<<gg,bb,smK128,stream>>>(
      Czb, W0t, b0, nullptr, nullptr, nullptr, nullptr, hA, nullptr);
  gemm_k<DH,   true,true,false,false, false,true,false><<<gg,bb,smK512,stream>>>(
      hA, Wm0t, bm, nullptr, nullptr, nullptr, nullptr, hB, nullptr);
  gemm_k<DH,   true,true,false,false, false,true,false><<<gg,bb,smK512,stream>>>(
      hB, Wm1t, bm+DH, nullptr, nullptr, nullptr, nullptr, hA, nullptr);
  // d = where(mask,0, h@Wl+bl) + task_emb
  gemm_k<DH,   true,false,true,false, false,true,false><<<gg,bb,smK512,stream>>>(
      hA, Wlt, bl, nullptr, maskF, te, nullptr, dbf, nullptr);

  // ---- attention block 0 ----
  gemm_k<DXCH, false,false,false,false, true,true,false><<<gg,bb,smK64,stream>>>(
      XDb, Wq0t, nullptr, nullptr, nullptr, nullptr, qF, qbf, nullptr);     // q0
  gemm_k<DXCH, false,false,false,false, false,true,false><<<gg,bb,smK64,stream>>>(
      XCb, Wk0t, nullptr, nullptr, nullptr, nullptr, nullptr, kbf, nullptr);// k0
  gemm_k<DH,   false,false,false,false, false,false,true><<<gg,bb,smK512,stream>>>(
      dbf, Wv0t, nullptr, nullptr, nullptr, nullptr, nullptr, nullptr, vT); // v0^T
  attn_k<<<ga, 256, ATTN_SMEM, stream>>>(qbf, kbf, vT, maskF, qF, hA);      // out0 in qF/hA
  // u0 = out0 + relu(out0@Wo0+bo0)  (bf16 -> hB, feeds q1)
  gemm_k<DH,   true,true,false,true, false,true,false><<<gg,bb,smK512,stream>>>(
      hA, Wo0t, bo0, qF, nullptr, nullptr, nullptr, hB, nullptr);

  // ---- attention block 1 ----
  gemm_k<DH,   false,false,false,false, true,true,false><<<gg,bb,smK512,stream>>>(
      hB, Wq1t, nullptr, nullptr, nullptr, nullptr, qF, qbf, nullptr);      // q1
  gemm_k<DXCH, false,false,false,false, false,true,false><<<gg,bb,smK64,stream>>>(
      XCb, Wk1t, nullptr, nullptr, nullptr, nullptr, nullptr, kbf, nullptr);// k1
  gemm_k<DH,   false,false,false,false, false,false,true><<<gg,bb,smK512,stream>>>(
      dbf, Wv1t, nullptr, nullptr, nullptr, nullptr, nullptr, nullptr, vT); // v1^T
  attn_k<<<ga, 256, ATTN_SMEM, stream>>>(qbf, kbf, vT, maskF, qF, hA);      // out1 in qF/hA
  // u1 = out1 + relu(out1@Wo1+bo1) -> final f32 output
  gemm_k<DH,   true,true,false,true, true,false,false><<<gg,bb,smK512,stream>>>(
      hA, Wo1t, bo1, qF, nullptr, nullptr, (float*)d_out, nullptr, nullptr);
}
